// Potential_90975997264409
// MI455X (gfx1250) — matrix operand order repaired and checked by reading, not compiled, not run on hardware
//
#include <hip/hip_runtime.h>
#include <hip/hip_bf16.h>

// ---------------------------------------------------------------------------
// Problem constants (from reference): B=2, S=2048, D=512, WIN=64, EPS=1e-5
// ---------------------------------------------------------------------------
#define BATCH 2
#define SEQ   2048
#define DIM   512
#define WIN   64
#define MTOT  (BATCH * SEQ)        // 4096 rows total
#define QTILES (MTOT / 16)         // 256 query tiles of 16 rows
#define KEYS_PER_TILE 96           // 6 key tiles of 16 cover [s0-64, s0+32)
#define PADS  (64 + SEQ + 16)      // padded seq length for transposed V (2128)
#define LN_EPS 1e-5f

#if __has_builtin(__builtin_amdgcn_sched_barrier)
#define SCHED_FENCE() __builtin_amdgcn_sched_barrier(0)
#else
#define SCHED_FENCE()
#endif

typedef __attribute__((ext_vector_type(16))) _Float16 v16h;
typedef __attribute__((ext_vector_type(8)))  _Float16 v8h;
typedef __attribute__((ext_vector_type(4)))  _Float16 v4h;
typedef __attribute__((ext_vector_type(8)))  float    v8f;

__device__ __forceinline__ int iclamp(int v, int lo, int hi) {
    return v < lo ? lo : (v > hi ? hi : v);
}

__device__ __forceinline__ v16h concat8(v8h a, v8h b) {
    return __builtin_shufflevector(a, b, 0, 1, 2, 3, 4, 5, 6, 7,
                                   8, 9, 10, 11, 12, 13, 14, 15);
}

// ---------------------------------------------------------------------------
// Fragment loaders (wave32); contiguous k order chosen by this code, same for A and B. All are two b128 loads.
//
// A (16x32 f16): lane L holds row M=L%16.
//   lo half-wave: K 0..15 contiguous ; hi half-wave: K 16..31
// B (32x16 f16): lane L holds col N=L%16.
//   lo half-wave: K 0..15 contiguous ; hi half-wave: K 16..31
// C/D (16x16 f32): lane L holds col N=L%16, VGPR r -> row r + (hiHalf ? 8:0)
// ---------------------------------------------------------------------------
__device__ __forceinline__ v16h load_a16x32(const _Float16* __restrict__ p,
                                            int ld, int r0, int c0) {
    int lane = threadIdx.x & 31;
    int m    = lane & 15;
    int hi   = lane >> 4;
    const _Float16* row = p + (size_t)(r0 + m) * ld + c0 + (hi << 4);
    v8h lo = *(const v8h*)row;
    v8h up = *(const v8h*)(row + 8);
    return concat8(lo, up);
}

// B element (k, n) = src[(n0+n)*ld + c0 + k]   (K-contiguous source rows)
__device__ __forceinline__ v16h load_b_kcontig(const _Float16* __restrict__ p,
                                               size_t ld, int n0, int c0) {
    int lane = threadIdx.x & 31;
    int n    = lane & 15;
    int hi   = lane >> 4;
    const _Float16* base = p + (size_t)(n0 + n) * ld + c0 + (hi << 4);
    v8h lo = *(const v8h*)base;
    v8h up = *(const v8h*)(base + 8);
    return concat8(lo, up);
}

// Same, but the per-lane source row is clamped (for K^T at batch edges).
__device__ __forceinline__ v16h load_b_kcontig_clamp(
    const _Float16* __restrict__ p, size_t ld, int row0, int c0,
    int rlo, int rhi) {
    int lane = threadIdx.x & 31;
    int n    = lane & 15;
    int hi   = lane >> 4;
    int row  = iclamp(row0 + n, rlo, rhi);
    const _Float16* base = p + (size_t)row * ld + c0 + (hi << 4);
    v8h lo = *(const v8h*)base;
    v8h up = *(const v8h*)(base + 8);
    return concat8(lo, up);
}

// ---------------------------------------------------------------------------
// Casts / transposes / zero-fill
// ---------------------------------------------------------------------------
__global__ void cast_f32_f16_v4(const float* __restrict__ src,
                                _Float16* __restrict__ dst, int n) {
    int i = (blockIdx.x * blockDim.x + threadIdx.x) * 4;
    if (i < n) {
        const float4 v = *(const float4*)(src + i);
        v4h h;
        h[0] = (_Float16)v.x; h[1] = (_Float16)v.y;
        h[2] = (_Float16)v.z; h[3] = (_Float16)v.w;
        *(v4h*)(dst + i) = h;
    }
}

// dst[n][k] = (f16) src[k][n]  for a 512x512 matrix (LDS-tiled)
__global__ __launch_bounds__(256)
void transpose_cast_512(const float* __restrict__ src,
                        _Float16* __restrict__ dst) {
    __shared__ _Float16 tile[32][33];
    int c0 = blockIdx.x * 32;
    int r0 = blockIdx.y * 32;
    int tx = threadIdx.x & 31;
    int ty = threadIdx.x >> 5;            // 0..7
#pragma unroll
    for (int i = ty; i < 32; i += 8)
        tile[i][tx] = (_Float16)src[(size_t)(r0 + i) * DIM + c0 + tx];
    __syncthreads();
#pragma unroll
    for (int i = ty; i < 32; i += 8)
        dst[(size_t)(c0 + i) * DIM + r0 + tx] = tile[tx][i];
}

__global__ void zero_f16(_Float16* __restrict__ p, int n) {
    int i = (blockIdx.x * blockDim.x + threadIdx.x) * 8;
    if (i < n) *(v8h*)(p + i) = v8h{};
}

// ---------------------------------------------------------------------------
// WMMA GEMM + bias; B operand given TRANSPOSED (BwT is N x K, K-contiguous).
// One wave computes a 16x64 output tile. All fragments of a K-step are loaded
// (scheduling fence) before the WMMA chain so the backend can emit partial
// s_wait_loadcnt values and overlap tail loads with leading WMMAs.
//   MODE 0: f16 row-major out      MODE 1: f32 row-major out
//   MODE 2: f16 transposed + edge-padded out (for V): vt[DIM][BATCH*PADS]
// ---------------------------------------------------------------------------
#define OUT_F16 0
#define OUT_F32 1
#define OUT_VT  2

template <int MODE, typename OutT>
__global__ __launch_bounds__(256)
void wmma_gemm_bias(const _Float16* __restrict__ A,    // M x K row-major
                    const _Float16* __restrict__ BwT,  // N x K row-major (W^T)
                    const float* __restrict__ bias,    // N
                    OutT* __restrict__ outp,
                    int M, int N, int K) {
    int wave = (blockIdx.x * blockDim.x + threadIdx.x) >> 5;
    int ntn  = N >> 6;
    int mt   = wave / ntn;
    int nt   = wave - mt * ntn;
    if (mt * 16 >= M) return;              // wave-uniform
    int r0 = mt * 16, n0 = nt * 64;

    v8f c[4] = {v8f{}, v8f{}, v8f{}, v8f{}};
    for (int kk = 0; kk < K; kk += 32) {
        v16h a = load_a16x32(A, K, r0, kk);
        v16h bfr[4];
#pragma unroll
        for (int j = 0; j < 4; ++j)
            bfr[j] = load_b_kcontig(BwT, K, n0 + j * 16, kk);
        SCHED_FENCE();                     // issue all loads before any WMMA
#pragma unroll
        for (int j = 0; j < 4; ++j)
            c[j] = __builtin_amdgcn_wmma_f32_16x16x32_f16(
                false, a, false, bfr[j], (short)0, c[j], false, false);
    }

    int lane = threadIdx.x & 31;
    int n    = lane & 15;
    int hi   = lane >> 4;

    if constexpr (MODE == OUT_VT) {
        int batch = r0 >> 11;              // r0 / SEQ
        int sl0   = r0 - batch * SEQ;
        const size_t ldvt = (size_t)BATCH * PADS;
        size_t cbase = (size_t)batch * PADS + 64 + sl0 + (hi << 3);
#pragma unroll
        for (int j = 0; j < 4; ++j) {
            float bj = bias[n0 + j * 16 + n];
            v8h pk;
#pragma unroll
            for (int r = 0; r < 8; ++r) pk[r] = (_Float16)(c[j][r] + bj);
            *(v8h*)((_Float16*)outp + (size_t)(n0 + j * 16 + n) * ldvt + cbase) = pk;
        }
    } else {
        size_t base = (size_t)(r0 + (hi << 3)) * N + n0 + n;
#pragma unroll
        for (int j = 0; j < 4; ++j) {
            float bj = bias[n0 + j * 16 + n];
#pragma unroll
            for (int r = 0; r < 8; ++r) {
                float val = c[j][r] + bj;
                if constexpr (MODE == OUT_F16)
                    outp[base + (size_t)r * N + j * 16] = (_Float16)val;
                else
                    outp[base + (size_t)r * N + j * 16] = val;
            }
        }
    }
}

// ---------------------------------------------------------------------------
// Sliding-window attention scores + softmax.
// One wave per 16-query tile; 6 key tiles of 16 starting at kbase = s0-64.
// All 6 K^T fragments of a K-step loaded (fence) before the WMMA chain.
// P[tile][16][96] f16 probabilities (masked = 0).
// ---------------------------------------------------------------------------
__global__ __launch_bounds__(256)
void attn_scores(const _Float16* __restrict__ q,   // [MTOT][DIM]
                 const _Float16* __restrict__ k,   // [MTOT][DIM]
                 _Float16* __restrict__ P) {       // [QTILES][16][96]
    int wave = (blockIdx.x * blockDim.x + threadIdx.x) >> 5;
    if (wave >= QTILES) return;                    // wave-uniform

    int s0      = wave * 16;
    int batch   = s0 >> 11;
    int sl0     = s0 - batch * SEQ;
    int kbase_l = sl0 - 64;
    int brow0   = batch * SEQ;
    int brow1   = brow0 + SEQ - 1;

    v8f sc[6] = {v8f{}, v8f{}, v8f{}, v8f{}, v8f{}, v8f{}};
    for (int ks = 0; ks < DIM; ks += 32) {
        v16h a = load_a16x32(q, DIM, s0, ks);      // loaded once per K-step
        v16h bfr[6];
#pragma unroll
        for (int kt = 0; kt < 6; ++kt)
            bfr[kt] = load_b_kcontig_clamp(k, DIM, brow0 + kbase_l + kt * 16,
                                           ks, brow0, brow1);
        SCHED_FENCE();                     // issue all loads before any WMMA
#pragma unroll
        for (int kt = 0; kt < 6; ++kt)
            sc[kt] = __builtin_amdgcn_wmma_f32_16x16x32_f16(
                false, a, false, bfr[kt], (short)0, sc[kt], false, false);
    }

    int lane = threadIdx.x & 31;
    int n    = lane & 15;
    int hi   = lane >> 4;
    const float scale = 0.044194173824159216f;     // 1/sqrt(512)

#pragma unroll
    for (int r = 0; r < 8; ++r) {
        int srow = sl0 + r + hi * 8;               // local query position
        float m = -1e30f;
#pragma unroll
        for (int kt = 0; kt < 6; ++kt) {
            int kl = kbase_l + kt * 16 + n;
            bool valid = (kl >= 0) && (kl <= srow) && (kl >= srow - (WIN - 1));
            float val  = valid ? sc[kt][r] * scale : -1e9f;
            sc[kt][r]  = val;
            m = fmaxf(m, val);
        }
#pragma unroll
        for (int off = 1; off < 16; off <<= 1)
            m = fmaxf(m, __shfl_xor(m, off, 32));

        float s = 0.0f;
#pragma unroll
        for (int kt = 0; kt < 6; ++kt) {
            float e = __expf(sc[kt][r] - m);
            sc[kt][r] = e;
            s += e;
        }
#pragma unroll
        for (int off = 1; off < 16; off <<= 1)
            s += __shfl_xor(s, off, 32);
        float inv = 1.0f / s;

        _Float16* prow = P + ((size_t)wave * 16 + r + hi * 8) * KEYS_PER_TILE;
#pragma unroll
        for (int kt = 0; kt < 6; ++kt)
            prow[kt * 16 + n] = (_Float16)(sc[kt][r] * inv);
    }
}

// ---------------------------------------------------------------------------
// out(16x512) = P(16x96) @ V[kbase:kbase+96, :] per query tile, using the
// transposed + padded V (no clamping needed; pads are zero, masked P is zero).
// One wave per (query tile, 64-col chunk): 2048 waves.
// ---------------------------------------------------------------------------
__global__ __launch_bounds__(256)
void attn_pv(const _Float16* __restrict__ P,        // [QTILES][16][96]
             const _Float16* __restrict__ vt,       // [DIM][BATCH*PADS]
             _Float16* __restrict__ attnout) {      // [MTOT][DIM]
    int wave = (blockIdx.x * blockDim.x + threadIdx.x) >> 5;
    int mt = wave >> 3;
    int nc = wave & 7;
    if (mt >= QTILES) return;                       // wave-uniform

    int s0      = mt * 16;
    int batch   = s0 >> 11;
    int sl0     = s0 - batch * SEQ;
    int n0      = nc * 64;
    const size_t ldvt = (size_t)BATCH * PADS;
    // padded column index of local key kbase_l+kc : batch*PADS + 64 + sl0-64+kc
    const size_t pbase = (size_t)batch * PADS + sl0;

    int lane = threadIdx.x & 31;
    int n    = lane & 15;
    int hi   = lane >> 4;

    const _Float16* Pt = P + (size_t)mt * 16 * KEYS_PER_TILE;

    v8f c[4] = {v8f{}, v8f{}, v8f{}, v8f{}};
#pragma unroll
    for (int kc = 0; kc < KEYS_PER_TILE; kc += 32) {
        v16h a = load_a16x32(Pt, KEYS_PER_TILE, 0, kc);
        v16h bfr[4];
#pragma unroll
        for (int j = 0; j < 4; ++j) {
            const _Float16* base =
                vt + (size_t)(n0 + j * 16 + n) * ldvt + pbase + kc + (hi << 4);
            bfr[j] = concat8(*(const v8h*)base, *(const v8h*)(base + 8));
        }
        SCHED_FENCE();                     // issue all loads before any WMMA
#pragma unroll
        for (int j = 0; j < 4; ++j)
            c[j] = __builtin_amdgcn_wmma_f32_16x16x32_f16(
                false, a, false, bfr[j], (short)0, c[j], false, false);
    }

    size_t base = (size_t)(s0 + (hi << 3)) * DIM + n0 + n;
#pragma unroll
    for (int j = 0; j < 4; ++j)
#pragma unroll
        for (int r = 0; r < 8; ++r)
            attnout[base + (size_t)r * DIM + j * 16] = (_Float16)c[j][r];
}

// ---------------------------------------------------------------------------
// Per-row: layernorm(read), layernorm(sence), sum, exact GELU.
// One 256-thread block per row.
// ---------------------------------------------------------------------------
__global__ __launch_bounds__(256)
void ln_add_gelu(const float* __restrict__ readb,   // [MTOT][DIM]
                 const float* __restrict__ sence,   // [MTOT][DIM]
                 const float* __restrict__ gamma,   // [DIM]
                 const float* __restrict__ beta,    // [DIM]
                 float* __restrict__ out) {         // [MTOT][DIM]
    __shared__ float sh[256];
    int row = blockIdx.x;
    int t   = threadIdx.x;
    const float* rp = readb + (size_t)row * DIM;
    const float* sp = sence + (size_t)row * DIM;

    float r0 = rp[t], r1 = rp[t + 256];
    float s0 = sp[t], s1 = sp[t + 256];

    float acc[4] = {r0 + r1, r0 * r0 + r1 * r1, s0 + s1, s0 * s0 + s1 * s1};
    float red[4];
#pragma unroll
    for (int i = 0; i < 4; ++i) {
        sh[t] = acc[i];
        __syncthreads();
        for (int off = 128; off > 0; off >>= 1) {
            if (t < off) sh[t] += sh[t + off];
            __syncthreads();
        }
        red[i] = sh[0];
        __syncthreads();
    }

    const float invD = 1.0f / (float)DIM;
    float mu_r  = red[0] * invD;
    float var_r = red[1] * invD - mu_r * mu_r;
    float mu_s  = red[2] * invD;
    float var_s = red[3] * invD - mu_s * mu_s;
    float inv_r = rsqrtf(var_r + LN_EPS);
    float inv_s = rsqrtf(var_s + LN_EPS);

    float* op = out + (size_t)row * DIM;
#pragma unroll
    for (int half = 0; half < 2; ++half) {
        int col = t + half * 256;
        float rv = half ? r1 : r0;
        float sv = half ? s1 : s0;
        float g = gamma[col], bta = beta[col];
        float hr = (rv - mu_r) * inv_r * g + bta;
        float hs = (sv - mu_s) * inv_s * g + bta;
        float x  = hr + hs;
        op[col]  = 0.5f * x * (1.0f + erff(x * 0.70710678118654752f));
    }
}

// ---------------------------------------------------------------------------
// Launch
// ---------------------------------------------------------------------------
extern "C" void kernel_launch(void* const* d_in, const int* in_sizes, int n_in,
                              void* d_out, int out_size, void* d_ws, size_t ws_size,
                              hipStream_t stream) {
    const float* x     = (const float*)d_in[0];
    const float* Wq    = (const float*)d_in[1];
    const float* bq    = (const float*)d_in[2];
    const float* Wk    = (const float*)d_in[3];
    const float* bk    = (const float*)d_in[4];
    const float* Wv    = (const float*)d_in[5];
    const float* bv    = (const float*)d_in[6];
    const float* Wo    = (const float*)d_in[7];
    const float* bo    = (const float*)d_in[8];
    const float* Wr    = (const float*)d_in[9];
    const float* br    = (const float*)d_in[10];
    const float* gamma = (const float*)d_in[11];
    const float* beta  = (const float*)d_in[12];
    float* out = (float*)d_out;

    // workspace carve-up
    char* w = (char*)d_ws;
    _Float16* xh   = (_Float16*)w; w += (size_t)MTOT * DIM * 2;     // 4 MB
    _Float16* wqt  = (_Float16*)w; w += (size_t)DIM * DIM * 2;      // W^T (f16)
    _Float16* wkt  = (_Float16*)w; w += (size_t)DIM * DIM * 2;
    _Float16* wvt  = (_Float16*)w; w += (size_t)DIM * DIM * 2;
    _Float16* wot  = (_Float16*)w; w += (size_t)DIM * DIM * 2;
    _Float16* wrt  = (_Float16*)w; w += (size_t)DIM * DIM * 2;
    _Float16* qh   = (_Float16*)w; w += (size_t)MTOT * DIM * 2;
    _Float16* kh   = (_Float16*)w; w += (size_t)MTOT * DIM * 2;
    _Float16* vtp  = (_Float16*)w; w += (size_t)DIM * BATCH * PADS * 2; // V^T padded
    _Float16* Pb   = (_Float16*)w; w += (size_t)QTILES * 16 * KEYS_PER_TILE * 2;
    _Float16* aoh  = (_Float16*)w; w += (size_t)MTOT * DIM * 2;
    float* r32     = (float*)w;    w += (size_t)MTOT * DIM * 4;     // sence pre-LN
    float* rd32    = (float*)w;    w += (size_t)MTOT * DIM * 4;     // read pre-LN

    const int nx  = MTOT * DIM;                    // 2097152
    const int nvt = DIM * BATCH * PADS;            // 2179072

    cast_f32_f16_v4<<<nx / 4 / 256, 256, 0, stream>>>(x, xh, nx);

    dim3 tgrid(DIM / 32, DIM / 32);
    transpose_cast_512<<<tgrid, 256, 0, stream>>>(Wq, wqt);
    transpose_cast_512<<<tgrid, 256, 0, stream>>>(Wk, wkt);
    transpose_cast_512<<<tgrid, 256, 0, stream>>>(Wv, wvt);
    transpose_cast_512<<<tgrid, 256, 0, stream>>>(Wo, wot);
    transpose_cast_512<<<tgrid, 256, 0, stream>>>(Wr, wrt);

    zero_f16<<<(nvt / 8 + 255) / 256, 256, 0, stream>>>(vtp, nvt);

    // projections: waves = (M/16)*(N/64) = 2048 -> 256 blocks of 8 waves
    const int gemm_blocks = (MTOT / 16) * (DIM / 64) * 32 / 256;
    wmma_gemm_bias<OUT_F16, _Float16><<<gemm_blocks, 256, 0, stream>>>(
        xh, wqt, bq, qh, MTOT, DIM, DIM);
    wmma_gemm_bias<OUT_F16, _Float16><<<gemm_blocks, 256, 0, stream>>>(
        xh, wkt, bk, kh, MTOT, DIM, DIM);
    wmma_gemm_bias<OUT_VT, _Float16><<<gemm_blocks, 256, 0, stream>>>(
        xh, wvt, bv, vtp, MTOT, DIM, DIM);
    wmma_gemm_bias<OUT_F32, float><<<gemm_blocks, 256, 0, stream>>>(
        xh, wrt, br, r32, MTOT, DIM, DIM);

    // attention
    attn_scores<<<QTILES * 32 / 256, 256, 0, stream>>>(qh, kh, Pb);
    attn_pv<<<QTILES * 8 * 32 / 256, 256, 0, stream>>>(Pb, vtp, aoh);

    // output projection -> f32
    wmma_gemm_bias<OUT_F32, float><<<gemm_blocks, 256, 0, stream>>>(
        aoh, wot, bo, rd32, MTOT, DIM, DIM);

    // layernorm x2 + add + exact gelu
    ln_add_gelu<<<MTOT, 256, 0, stream>>>(rd32, r32, gamma, beta, out);
}
